// GAT_50594714746950
// MI455X (gfx1250) — compile-verified, matrix operand order checked by reading, not run on hardware
//
#include <hip/hip_runtime.h>
#include <hip/hip_bf16.h>

#define DIM 768
#define BM  128
#define BN  128
#define BK  32
#define LDSS 40   // bf16 elements per LDS row; 80B stride, 16B aligned

typedef __bf16 v16bf __attribute__((ext_vector_type(16)));
typedef float  v8f   __attribute__((ext_vector_type(8)));

// ---------------------------------------------------------------------------
// fp32 -> bf16 conversion (plain)
// ---------------------------------------------------------------------------
__global__ void gat_cvt_bf16(const float* __restrict__ src, __bf16* __restrict__ dst, int n) {
    int i = blockIdx.x * blockDim.x + threadIdx.x;
    if (i < n) dst[i] = (__bf16)src[i];
}

// fp32 -> bf16 with transpose:  dst[n*DIM + k] = (bf16) src[k*DIM + n]
__global__ void gat_cvt_bf16_T(const float* __restrict__ src, __bf16* __restrict__ dst) {
    int i = blockIdx.x * blockDim.x + threadIdx.x;   // coalesced read of src
    if (i >= DIM * DIM) return;
    int k = i / DIM;
    int n = i - k * DIM;
    dst[(size_t)n * DIM + k] = (__bf16)src[i];
}

// ---------------------------------------------------------------------------
// async global -> LDS staging (CDNA5 ASYNCcnt path)
// ---------------------------------------------------------------------------
__device__ __forceinline__ void async_b128(unsigned lds_off, const void* gaddr) {
    asm volatile("global_load_async_to_lds_b128 %0, %1, off"
                 :: "v"(lds_off), "v"(gaddr) : "memory");
}
__device__ __forceinline__ void wait_async0() {
    asm volatile("s_wait_asynccnt 0x0" ::: "memory");
}

// one 16B chunk pair (A row + Bt row) per thread, 2 passes -> 128x32 each
__device__ __forceinline__ void gemm_stage(
    const __bf16* __restrict__ A, const __bf16* __restrict__ Bt, int M,
    int bm, int bn, int k0, int tid, __bf16* AsBuf, __bf16* BsBuf)
{
    #pragma unroll
    for (int c = 0; c < 2; ++c) {
        int linear = tid + c * 256;        // 0..511
        int r  = linear >> 2;              // 0..127
        int kg = (linear & 3) * 8;         // 0,8,16,24
        int ga = bm + r; ga = (ga < M) ? ga : (M - 1);   // clamp (rows >= M unused)
        async_b128((unsigned)(uintptr_t)(AsBuf + r * LDSS + kg),
                   A + (size_t)ga * DIM + k0 + kg);
        async_b128((unsigned)(uintptr_t)(BsBuf + r * LDSS + kg),
                   Bt + (size_t)(bn + r) * DIM + k0 + kg);
    }
}

// ---------------------------------------------------------------------------
// WMMA GEMM: C[M x 768] = A[M x 768](bf16) * W[768 x 768], Bt = W^T (bf16)
// block tile 128x128, 8 waves, wave tile 32x64 (2x4 wmma), double-buffered LDS
// ---------------------------------------------------------------------------
__device__ __forceinline__ v16bf load_frag_row(const __bf16* rowbase, int lane) {
    // 16-bit A/B fragment per CDNA5 ISA: base8 = (lane/16)*8;
    // elems 0-7 -> K = base8..base8+7 ; elems 8-15 -> K = 16+base8..16+base8+7
    int b8 = (lane >> 4) * 8;
    v16bf f;
    uint4* p = reinterpret_cast<uint4*>(&f);
    p[0] = *reinterpret_cast<const uint4*>(rowbase + b8);
    p[1] = *reinterpret_cast<const uint4*>(rowbase + 16 + b8);
    return f;
}

__global__ __launch_bounds__(256) void gat_gemm_bf16(
    const __bf16* __restrict__ A, const __bf16* __restrict__ Bt,
    float* __restrict__ C, int M)
{
    __shared__ __align__(16) __bf16 As[2][BM * LDSS];
    __shared__ __align__(16) __bf16 Bs[2][BN * LDSS];   // Bs[n][k] from Bt rows

    const int tid  = threadIdx.x;
    const int lane = tid & 31;
    const int wave = tid >> 5;
    const int wm   = wave & 3;    // 4 waves along M  -> 32 rows each
    const int wn   = wave >> 2;   // 2 waves along N  -> 64 cols each
    const int bm   = blockIdx.x * BM;
    const int bn   = blockIdx.y * BN;

    v8f acc[2][4];
    #pragma unroll
    for (int i = 0; i < 2; ++i)
        #pragma unroll
        for (int j = 0; j < 4; ++j) { v8f z = {}; acc[i][j] = z; }

    gemm_stage(A, Bt, M, bm, bn, 0, tid, As[0], Bs[0]);

    const int mr = lane & 15;
    int buf = 0;
    for (int k0 = 0; k0 < DIM; k0 += BK, buf ^= 1) {
        wait_async0();
        __syncthreads();
        if (k0 + BK < DIM)
            gemm_stage(A, Bt, M, bm, bn, k0 + BK, tid, As[buf ^ 1], Bs[buf ^ 1]);

        v16bf afrag[2], bfrag[4];
        #pragma unroll
        for (int i = 0; i < 2; ++i)
            afrag[i] = load_frag_row(&As[buf][(wm * 32 + i * 16 + mr) * LDSS], lane);
        #pragma unroll
        for (int j = 0; j < 4; ++j)
            bfrag[j] = load_frag_row(&Bs[buf][(wn * 64 + j * 16 + mr) * LDSS], lane);

        #pragma unroll
        for (int i = 0; i < 2; ++i)
            #pragma unroll
            for (int j = 0; j < 4; ++j)
                acc[i][j] = __builtin_amdgcn_wmma_f32_16x16x32_bf16(
                    false, afrag[i], false, bfrag[j], (short)0, acc[i][j], false, false);
    }

    // store D: VGPR r, lanes 0-15 -> M=r ; lanes 16-31 -> M=8+r ; N = lane%16
    int col16 = lane & 15;
    int rbase = (lane >> 4) * 8;
    #pragma unroll
    for (int i = 0; i < 2; ++i) {
        #pragma unroll
        for (int j = 0; j < 4; ++j) {
            int col = bn + wn * 64 + j * 16 + col16;
            #pragma unroll
            for (int r = 0; r < 8; ++r) {
                int row = bm + wm * 32 + i * 16 + rbase + r;
                if (row < M) C[(size_t)row * DIM + col] = acc[i][j][r];
            }
        }
    }
}

// ---------------------------------------------------------------------------
// a_src[n] = dot(h[n], att_src);  a_dst[n] = dot(h[n], att_dst)
// ---------------------------------------------------------------------------
__global__ __launch_bounds__(256) void gat_rowdot(
    const float* __restrict__ h, const float* __restrict__ ws_,
    const float* __restrict__ wd_, float* __restrict__ a_s, float* __restrict__ a_d)
{
    __shared__ float ss[256], sd[256];
    int n = blockIdx.x;
    const float* row = h + (size_t)n * DIM;
    float ps = 0.f, pd = 0.f;
    for (int j = threadIdx.x; j < DIM; j += 256) {
        float v = row[j];
        ps += v * ws_[j];
        pd += v * wd_[j];
    }
    ss[threadIdx.x] = ps; sd[threadIdx.x] = pd;
    __syncthreads();
    for (int s = 128; s > 0; s >>= 1) {
        if ((int)threadIdx.x < s) {
            ss[threadIdx.x] += ss[threadIdx.x + s];
            sd[threadIdx.x] += sd[threadIdx.x + s];
        }
        __syncthreads();
    }
    if (threadIdx.x == 0) { a_s[n] = ss[0]; a_d[n] = sd[0]; }
}

// order-preserving float<->uint encoding for atomicMax-based segment max
__device__ __forceinline__ unsigned enc_f32(float f) {
    unsigned b = __float_as_uint(f);
    return b ^ (unsigned)(((int)b >> 31) | (int)0x80000000);
}
__device__ __forceinline__ float dec_f32(unsigned u) {
    unsigned b = (u & 0x80000000u) ? (u ^ 0x80000000u) : ~u;
    return __uint_as_float(b);
}

// ---------------------------------------------------------------------------
// edge pass 1: raw leaky-relu logits + segment max (atomicMax on encoded u32)
// ---------------------------------------------------------------------------
__global__ void gat_edge_max(const int* __restrict__ src, const int* __restrict__ dst,
                             int E, int NN, const float* __restrict__ a_s,
                             const float* __restrict__ a_d,
                             float* __restrict__ eraw, unsigned* __restrict__ menc)
{
    int e = blockIdx.x * blockDim.x + threadIdx.x;
    int EE = E + NN;
    if (e >= EE) return;
    int s = (e < E) ? src[e] : (e - E);
    int d = (e < E) ? dst[e] : (e - E);
    float v = a_s[s] + a_d[d];
    v = (v > 0.f) ? v : 0.2f * v;       // leaky_relu, slope 0.2
    eraw[e] = v;
    atomicMax(&menc[d], enc_f32(v));
}

// ---------------------------------------------------------------------------
// edge pass 2: exp(e - max) + segment sum of denominators
// ---------------------------------------------------------------------------
__global__ void gat_edge_exp(const int* __restrict__ dst, int E, int NN,
                             const float* __restrict__ eraw,
                             const unsigned* __restrict__ menc,
                             float* __restrict__ w, float* __restrict__ denom)
{
    int e = blockIdx.x * blockDim.x + threadIdx.x;
    int EE = E + NN;
    if (e >= EE) return;
    int d = (e < E) ? dst[e] : (e - E);
    float m = dec_f32(menc[d]);
    float ww = __expf(eraw[e] - m);
    w[e] = ww;
    __hip_atomic_fetch_add(&denom[d], ww, __ATOMIC_RELAXED, __HIP_MEMORY_SCOPE_AGENT);
}

// ---------------------------------------------------------------------------
// edge pass 3: out[dst] += alpha * h[src]   (one wave32 per edge)
// ---------------------------------------------------------------------------
__global__ __launch_bounds__(256) void gat_edge_agg(
    const int* __restrict__ src, const int* __restrict__ dst, int E, int NN,
    const float* __restrict__ w, const float* __restrict__ denom,
    const float* __restrict__ h, float* __restrict__ out)
{
    int lane = threadIdx.x & 31;
    int wid  = blockIdx.x * (blockDim.x >> 5) + (threadIdx.x >> 5);
    int EE = E + NN;
    if (wid >= EE) return;
    int s = (wid < E) ? src[wid] : (wid - E);
    int d = (wid < E) ? dst[wid] : (wid - E);
    float alpha = w[wid] / denom[d];
    const float* hrow = h + (size_t)s * DIM;
    float*       orow = out + (size_t)d * DIM;
    __builtin_prefetch(hrow + lane * 4, 0, 1);   // global_prefetch_b8
    #pragma unroll
    for (int j = lane; j < DIM; j += 32) {
        __hip_atomic_fetch_add(&orow[j], alpha * hrow[j],
                               __ATOMIC_RELAXED, __HIP_MEMORY_SCOPE_AGENT);
    }
}

// ---------------------------------------------------------------------------
// epilogues
// ---------------------------------------------------------------------------
__global__ void gat_relu_bf16(const float* __restrict__ acc, const float* __restrict__ bias,
                              __bf16* __restrict__ xb, int total)
{
    int i = blockIdx.x * blockDim.x + threadIdx.x;
    if (i >= total) return;
    float v = acc[i] + bias[i % DIM];
    v = (v > 0.f) ? v : 0.f;
    xb[i] = (__bf16)v;
}

__global__ void gat_final(const float* __restrict__ acc, const float* __restrict__ bias,
                          float* __restrict__ out, int total)
{
    int i = blockIdx.x * blockDim.x + threadIdx.x;
    if (i >= total) return;
    out[i] = acc[i] + bias[i % DIM];
}

// ---------------------------------------------------------------------------
// driver
// ---------------------------------------------------------------------------
extern "C" void kernel_launch(void* const* d_in, const int* in_sizes, int n_in,
                              void* d_out, int out_size, void* d_ws, size_t ws_size,
                              hipStream_t stream) {
    const float* x        = (const float*)d_in[0];
    const int*   ei       = (const int*)d_in[1];   // [2, E] (JAX default int32)
    const float* W1       = (const float*)d_in[2];
    const float* att_s1   = (const float*)d_in[3];
    const float* att_d1   = (const float*)d_in[4];
    const float* b1       = (const float*)d_in[5];
    const float* W2       = (const float*)d_in[6];
    const float* att_s2   = (const float*)d_in[7];
    const float* att_d2   = (const float*)d_in[8];
    const float* b2       = (const float*)d_in[9];

    const int N  = in_sizes[0] / DIM;     // 10000
    const int E  = in_sizes[1] / 2;       // 100000
    const int EE = E + N;                 // with self-loops
    const int* src = ei;
    const int* dst = ei + E;

    // workspace carve-up (256B aligned)
    char* ws = (char*)d_ws;
    size_t off = 0;
    auto carve = [&](size_t bytes) -> void* {
        void* p = ws + off;
        off = (off + bytes + 255) & ~(size_t)255;
        return p;
    };
    float*    h      = (float*)   carve((size_t)N * DIM * sizeof(float));
    float*    acc    = (float*)   carve((size_t)N * DIM * sizeof(float));
    __bf16*   xb     = (__bf16*)  carve((size_t)N * DIM * sizeof(__bf16));
    __bf16*   Wt     = (__bf16*)  carve((size_t)DIM * DIM * sizeof(__bf16));  // W^T bf16
    float*    a_s    = (float*)   carve((size_t)N * sizeof(float));
    float*    a_d    = (float*)   carve((size_t)N * sizeof(float));
    unsigned* menc   = (unsigned*)carve((size_t)N * sizeof(unsigned));
    float*    denom  = (float*)   carve((size_t)N * sizeof(float));
    float*    eraw   = (float*)   carve((size_t)EE * sizeof(float));
    float*    wexp   = (float*)   carve((size_t)EE * sizeof(float));

    const int total   = N * DIM;
    const int tBlocks = (total + 255) / 256;
    const int wBlocks = (DIM * DIM + 255) / 256;
    const int eBlocks = (EE + 255) / 256;
    const int aBlocks = (EE + 7) / 8;               // 8 waves (edges) per block
    dim3 gemmGrid((N + BM - 1) / BM, DIM / BN);

    // ---------------- layer 1 ----------------
    gat_cvt_bf16  <<<tBlocks, 256, 0, stream>>>(x, xb, total);
    gat_cvt_bf16_T<<<wBlocks, 256, 0, stream>>>(W1, Wt);
    hipMemsetAsync(acc,   0, (size_t)total * sizeof(float), stream);
    hipMemsetAsync(menc,  0, (size_t)N * sizeof(unsigned), stream);
    hipMemsetAsync(denom, 0, (size_t)N * sizeof(float), stream);

    gat_gemm_bf16<<<gemmGrid, 256, 0, stream>>>(xb, Wt, h, N);
    gat_rowdot  <<<N, 256, 0, stream>>>(h, att_s1, att_d1, a_s, a_d);
    gat_edge_max<<<eBlocks, 256, 0, stream>>>(src, dst, E, N, a_s, a_d, eraw, menc);
    gat_edge_exp<<<eBlocks, 256, 0, stream>>>(dst, E, N, eraw, menc, wexp, denom);
    gat_edge_agg<<<aBlocks, 256, 0, stream>>>(src, dst, E, N, wexp, denom, h, acc);
    gat_relu_bf16<<<tBlocks, 256, 0, stream>>>(acc, b1, xb, total);  // next layer's input

    // ---------------- layer 2 ----------------
    gat_cvt_bf16_T<<<wBlocks, 256, 0, stream>>>(W2, Wt);
    hipMemsetAsync(acc,   0, (size_t)total * sizeof(float), stream);
    hipMemsetAsync(menc,  0, (size_t)N * sizeof(unsigned), stream);
    hipMemsetAsync(denom, 0, (size_t)N * sizeof(float), stream);

    gat_gemm_bf16<<<gemmGrid, 256, 0, stream>>>(xb, Wt, h, N);
    gat_rowdot  <<<N, 256, 0, stream>>>(h, att_s2, att_d2, a_s, a_d);
    gat_edge_max<<<eBlocks, 256, 0, stream>>>(src, dst, E, N, a_s, a_d, eraw, menc);
    gat_edge_exp<<<eBlocks, 256, 0, stream>>>(dst, E, N, eraw, menc, wexp, denom);
    gat_edge_agg<<<aBlocks, 256, 0, stream>>>(src, dst, E, N, wexp, denom, h, acc);
    gat_final   <<<tBlocks, 256, 0, stream>>>(acc, b2, (float*)d_out, total);
}
